// PyGEGNNDecoderMSE_15917148799551
// MI455X (gfx1250) — matrix operand order repaired and checked by reading, not compiled, not run on hardware
//
#include <hip/hip_runtime.h>

// ---------------------------------------------------------------------------
// EGNN decoder, fused for MI455X (gfx1250, wave32, WMMA f32_16x16x32_f16).
// B=512, NA=58, LAT=64, HID=128, AF=10, L=2, N=29696, E=475136.
// All GEMMs run on v_wmma_f32_16x16x32_f16 with f32 accumulation; activations
// and weights are carried in f16 (values are O(1) after silu, so f16 range is
// safe); segment sums use f32 global atomics.
// ---------------------------------------------------------------------------

#define BB    512
#define NAa   58
#define LATd  64
#define HIDd  128
#define AFd   10
#define NLay  2
#define NNod  (BB * NAa)      // 29696
#define NEdg  (NNod * 16)     // 475136

typedef __attribute__((ext_vector_type(16))) _Float16 v16h;
typedef __attribute__((ext_vector_type(8)))  _Float16 h8;
typedef __attribute__((ext_vector_type(8)))  float    v8f;

__device__ __forceinline__ float silu_f(float x) { return x / (1.0f + __expf(-x)); }

__device__ __forceinline__ v8f wmma16(v16h a, v16h b, v8f c) {
  // (neg_a, A, neg_b, B, c_mod, C, reuse_a, reuse_b)
  return __builtin_amdgcn_wmma_f32_16x16x32_f16(false, a, false, b, (short)0, c, false, false);
}

// A fragment (16x32 f16 tile of the M x K activation matrix) from LDS.
// ISA layout: lanes 0-15 row M=lane hold K={kb..kb+7, kb+16..kb+23} with kb=0;
// lanes 16-31 same rows with kb=8.
__device__ __forceinline__ v16h load_a_frag(const _Float16* base, int ldk, int kstep, int lane) {
  const int m  = lane & 15;
  const int kb = kstep * 32 + ((lane & 16) ? 8 : 0);
  const h8* p  = (const h8*)(base + m * ldk + kb);
  h8 lo = p[0];
  h8 hi = p[2];   // +16 halves
  v16h a;
#pragma unroll
  for (int i = 0; i < 8; ++i) { a[i] = lo[i]; a[8 + i] = hi[i]; }
  return a;
}

// B fragment (32x16 f16 tile) from pre-transposed weights wt[N][K]:
// lanes 0-15 column N=n0+lane hold K={kb..kb+7, kb+16..kb+23} with kb=ks*32; lanes 16-31 same with kb+8.
__device__ __forceinline__ v16h load_b_frag(const _Float16* wt, int K, int n0, int kstep, int lane) {
  const int n  = n0 + (lane & 15);
  const int kb = kstep * 32 + ((lane & 16) ? 8 : 0);
  const h8* p  = (const h8*)(wt + (size_t)n * K + kb);
  h8 lo = p[0], hi = p[2];
  v16h b;
#pragma unroll
  for (int i = 0; i < 8; ++i) { b[i] = lo[i]; b[8 + i] = hi[i]; }
  return b;
}

// ---------------------------------------------------------------------------
// prep kernels
// ---------------------------------------------------------------------------
__global__ void zero_kernel(float* __restrict__ p, size_t n) {
  size_t i = (size_t)blockIdx.x * 256 + threadIdx.x;
  if (i < n) p[i] = 0.0f;
}

// w: [K][Nc] f32 row-major  ->  wt: [Nc][K] f16 (transposed for B-fragments)
__global__ void convert_wt(const float* __restrict__ w, _Float16* __restrict__ wt, int K, int Nc) {
  int idx = blockIdx.x * 256 + threadIdx.x;
  if (idx >= K * Nc) return;
  int k = idx / Nc, n = idx % Nc;
  wt[(size_t)n * K + k] = (_Float16)w[(size_t)k * Nc + n];
}

__global__ void deg_kernel(const int* __restrict__ row, float* __restrict__ deg) {
  int e = blockIdx.x * 256 + threadIdx.x;
  if (e < NEdg) atomicAdd(&deg[row[e]], 1.0f);
}

// h = broadcast(z @ latent_w + latent_b) + atom_types @ atom_w + atom_b ; pos init
__global__ __launch_bounds__(128) void init_kernel(
    const float* __restrict__ z, const float* __restrict__ at,
    const float* __restrict__ lw, const float* __restrict__ lb,
    const float* __restrict__ aw, const float* __restrict__ ab,
    const float* __restrict__ init_coords,
    float* __restrict__ h, _Float16* __restrict__ h16, float* __restrict__ pos) {
  const int node = blockIdx.x;
  const int c = threadIdx.x;                 // 0..127
  const int b = node / NAa, a = node % NAa;
  float acc = lb[c] + ab[c];
  for (int k = 0; k < LATd; ++k) acc += z[b * LATd + k] * lw[k * HIDd + c];
  for (int k = 0; k < AFd;  ++k) acc += at[node * AFd + k] * aw[k * HIDd + c];
  h[(size_t)node * HIDd + c] = acc;
  h16[(size_t)node * HIDd + c] = (_Float16)acc;
  if (c < 3) pos[node * 3 + c] = init_coords[a * 3 + c];
}

// ---------------------------------------------------------------------------
// fused edge kernel: 16 edges / block, 8 waves each owning one 16-col tile.
// GEMM1 (K=257 via rank-1 d2 fold + 8 WMMA), GEMM2 (4 WMMA), message agg,
// GEMM3 coord MLP (4 WMMA), coord scalar + position accumulation.
// ---------------------------------------------------------------------------
__global__ __launch_bounds__(256) void edge_kernel(
    const _Float16* __restrict__ h16, const float* __restrict__ pos,
    const int* __restrict__ row, const int* __restrict__ col,
    const _Float16* __restrict__ ew1_t, const float* __restrict__ ew1_last,
    const float* __restrict__ eb1,
    const _Float16* __restrict__ ew2_t, const float* __restrict__ eb2,
    const _Float16* __restrict__ cw1_t, const float* __restrict__ cb1,
    const float* __restrict__ cw2,
    float* __restrict__ agg, float* __restrict__ upd) {
  __shared__ __align__(16) _Float16 sA[16][256];   // [xi | xj] f16
  __shared__ __align__(16) _Float16 sH1[16][128];  // silu(ef@ew1+eb1)
  __shared__ __align__(16) _Float16 sH2[16][128];  // m
  __shared__ float sCW[16][128];                   // silu(m@cw1+cb1)
  __shared__ float sd2[16];
  __shared__ float srel[16][3];
  __shared__ int   srow[16];
  __shared__ int   scol[16];

  const int tid = threadIdx.x;
  const int lane = tid & 31;
  const int wv = tid >> 5;
  const int e0 = blockIdx.x * 16;

  if (tid < 16) {
    int r = row[e0 + tid], c = col[e0 + tid];
    srow[tid] = r; scol[tid] = c;
    float d2 = 0.0f;
#pragma unroll
    for (int i = 0; i < 3; ++i) {
      float d = pos[r * 3 + i] - pos[c * 3 + i];
      srel[tid][i] = d; d2 += d * d;
    }
    sd2[tid] = fminf(fmaxf(d2, 1e-6f), 1e6f);
  }
  __syncthreads();

  for (int i = tid; i < 16 * 256; i += 256) {
    int m = i >> 8, k = i & 255;
    int src = (k < HIDd) ? srow[m] : scol[m];
    sA[m][k] = h16[(size_t)src * HIDd + (k & (HIDd - 1))];
  }
  __syncthreads();

  const int n0 = wv * 16;
  const int n  = n0 + (lane & 15);
  const int mb = (lane & 16) ? 8 : 0;

  // ---- GEMM1: ef[16x257] @ ew1 ; K=257th element folded into C init ----
  v8f acc;
  {
    float bias = eb1[n], wl = ew1_last[n];
#pragma unroll
    for (int r = 0; r < 8; ++r) acc[r] = bias + sd2[mb + r] * wl;
  }
#pragma unroll
  for (int ks = 0; ks < 8; ++ks) {
    v16h a = load_a_frag(&sA[0][0], 256, ks, lane);
    v16h b = load_b_frag(ew1_t, 256, n0, ks, lane);
    acc = wmma16(a, b, acc);
  }
#pragma unroll
  for (int r = 0; r < 8; ++r) sH1[mb + r][n] = (_Float16)silu_f(acc[r]);
  __syncthreads();

  // ---- GEMM2: -> m ----
  {
    float bias = eb2[n];
#pragma unroll
    for (int r = 0; r < 8; ++r) acc[r] = bias;
  }
#pragma unroll
  for (int ks = 0; ks < 4; ++ks) {
    v16h a = load_a_frag(&sH1[0][0], 128, ks, lane);
    v16h b = load_b_frag(ew2_t, 128, n0, ks, lane);
    acc = wmma16(a, b, acc);
  }
  float mv[8];
#pragma unroll
  for (int r = 0; r < 8; ++r) {
    mv[r] = silu_f(acc[r]);
    sH2[mb + r][n] = (_Float16)mv[r];
  }
  // segment_sum(m, row)
#pragma unroll
  for (int r = 0; r < 8; ++r)
    atomicAdd(&agg[(size_t)srow[mb + r] * HIDd + n], mv[r]);
  __syncthreads();

  // ---- GEMM3: coord MLP layer 1 ----
  {
    float bias = cb1[n];
#pragma unroll
    for (int r = 0; r < 8; ++r) acc[r] = bias;
  }
#pragma unroll
  for (int ks = 0; ks < 4; ++ks) {
    v16h a = load_a_frag(&sH2[0][0], 128, ks, lane);
    v16h b = load_b_frag(cw1_t, 128, n0, ks, lane);
    acc = wmma16(a, b, acc);
  }
#pragma unroll
  for (int r = 0; r < 8; ++r) sCW[mb + r][n] = silu_f(acc[r]);
  __syncthreads();

  // ---- per-edge coord scalar: clip(cwh @ cw2) ; segment_sum(cw*rel, row) ----
  if (tid < 16) {
    float s = 0.0f;
    for (int k = 0; k < HIDd; ++k) s += sCW[tid][k] * cw2[k];
    s = fminf(fmaxf(s, -1.0f), 1.0f);
#pragma unroll
    for (int i = 0; i < 3; ++i)
      atomicAdd(&upd[(size_t)srow[tid] * 3 + i], s * srel[tid][i]);
  }
}

// ---------------------------------------------------------------------------
// node kernel: 16 nodes / block. concat(h,agg)[16x256] @ nw1 -> silu -> @ nw2,
// then LayerNorm, write h/h16 in place.
// ---------------------------------------------------------------------------
__global__ __launch_bounds__(256) void node_kernel(
    const _Float16* __restrict__ h16_in, const float* __restrict__ agg,
    const _Float16* __restrict__ nw1_t, const float* __restrict__ nb1,
    const _Float16* __restrict__ nw2_t, const float* __restrict__ nb2,
    const float* __restrict__ lg, const float* __restrict__ lb,
    float* __restrict__ h, _Float16* __restrict__ h16_out) {
  __shared__ __align__(16) _Float16 sA[16][256];
  __shared__ __align__(16) _Float16 sH1[16][128];
  __shared__ float sHN[16][128];
  __shared__ float smu[16], srs[16];

  const int tid = threadIdx.x;
  const int lane = tid & 31;
  const int wv = tid >> 5;
  const int node0 = blockIdx.x * 16;

  for (int i = tid; i < 16 * 256; i += 256) {
    int m = i >> 8, k = i & 255;
    int node = node0 + m;
    sA[m][k] = (k < HIDd) ? h16_in[(size_t)node * HIDd + k]
                          : (_Float16)agg[(size_t)node * HIDd + (k - HIDd)];
  }
  __syncthreads();

  const int n0 = wv * 16;
  const int n  = n0 + (lane & 15);
  const int mb = (lane & 16) ? 8 : 0;

  v8f acc;
  {
    float bias = nb1[n];
#pragma unroll
    for (int r = 0; r < 8; ++r) acc[r] = bias;
  }
#pragma unroll
  for (int ks = 0; ks < 8; ++ks) {
    v16h a = load_a_frag(&sA[0][0], 256, ks, lane);
    v16h b = load_b_frag(nw1_t, 256, n0, ks, lane);
    acc = wmma16(a, b, acc);
  }
#pragma unroll
  for (int r = 0; r < 8; ++r) sH1[mb + r][n] = (_Float16)silu_f(acc[r]);
  __syncthreads();

  {
    float bias = nb2[n];
#pragma unroll
    for (int r = 0; r < 8; ++r) acc[r] = bias;
  }
#pragma unroll
  for (int ks = 0; ks < 4; ++ks) {
    v16h a = load_a_frag(&sH1[0][0], 128, ks, lane);
    v16h b = load_b_frag(nw2_t, 128, n0, ks, lane);
    acc = wmma16(a, b, acc);
  }
#pragma unroll
  for (int r = 0; r < 8; ++r) sHN[mb + r][n] = acc[r];
  __syncthreads();

  if (tid < 16) {
    float mu = 0.0f;
    for (int k = 0; k < HIDd; ++k) mu += sHN[tid][k];
    mu *= (1.0f / HIDd);
    float var = 0.0f;
    for (int k = 0; k < HIDd; ++k) { float d = sHN[tid][k] - mu; var += d * d; }
    var *= (1.0f / HIDd);
    smu[tid] = mu; srs[tid] = rsqrtf(var + 1e-5f);
  }
  __syncthreads();

  for (int i = tid; i < 16 * HIDd; i += 256) {
    int m = i >> 7, k = i & (HIDd - 1);
    int node = node0 + m;
    float v = (sHN[m][k] - smu[m]) * srs[m] * lg[k] + lb[k];
    h[(size_t)node * HIDd + k] = v;
    h16_out[(size_t)node * HIDd + k] = (_Float16)v;
  }
}

__global__ void pos_update(float* __restrict__ pos, const float* __restrict__ upd,
                           const float* __restrict__ deg) {
  int i = blockIdx.x * 256 + threadIdx.x;
  if (i < NNod * 3) {
    int node = i / 3;
    pos[i] += upd[i] / (deg[node] + 1e-6f);
  }
}

// ---------------------------------------------------------------------------
// head: silu(h @ head_w1 + b1) @ head_w2 + b2 + pos
// ---------------------------------------------------------------------------
__global__ __launch_bounds__(256) void head_kernel(
    const _Float16* __restrict__ h16, const float* __restrict__ pos,
    const _Float16* __restrict__ hw1_t, const float* __restrict__ hb1,
    const float* __restrict__ hw2, const float* __restrict__ hb2,
    float* __restrict__ out) {
  __shared__ __align__(16) _Float16 sA[16][128];
  __shared__ float sH1[16][64];

  const int tid = threadIdx.x;
  const int lane = tid & 31;
  const int wv = tid >> 5;
  const int node0 = blockIdx.x * 16;

  for (int i = tid; i < 16 * HIDd; i += 256) {
    int m = i >> 7, k = i & (HIDd - 1);
    sA[m][k] = h16[(size_t)(node0 + m) * HIDd + k];
  }
  __syncthreads();

  if (wv < 4) {                                   // 64 output cols -> 4 tiles
    const int n0 = wv * 16;
    const int n  = n0 + (lane & 15);
    const int mb = (lane & 16) ? 8 : 0;
    v8f acc;
    float bias = hb1[n];
#pragma unroll
    for (int r = 0; r < 8; ++r) acc[r] = bias;
#pragma unroll
    for (int ks = 0; ks < 4; ++ks) {
      v16h a = load_a_frag(&sA[0][0], 128, ks, lane);
      v16h b = load_b_frag(hw1_t, 128, n0, ks, lane);
      acc = wmma16(a, b, acc);
    }
#pragma unroll
    for (int r = 0; r < 8; ++r) sH1[mb + r][n] = silu_f(acc[r]);
  }
  __syncthreads();

  if (tid < 48) {
    int m = tid / 3, j = tid % 3;
    float d = hb2[j];
    for (int k = 0; k < 64; ++k) d += sH1[m][k] * hw2[k * 3 + j];
    out[(size_t)(node0 + m) * 3 + j] = pos[(node0 + m) * 3 + j] + d;
  }
}

// ---------------------------------------------------------------------------
extern "C" void kernel_launch(void* const* d_in, const int* in_sizes, int n_in,
                              void* d_out, int out_size, void* d_ws, size_t ws_size,
                              hipStream_t stream) {
  (void)in_sizes; (void)n_in; (void)out_size; (void)ws_size;
  const float* z           = (const float*)d_in[0];
  const float* atom_types  = (const float*)d_in[1];
  const int*   edge_index  = (const int*)d_in[2];
  const float* init_coords = (const float*)d_in[3];
  const float* latent_w    = (const float*)d_in[4];
  const float* latent_b    = (const float*)d_in[5];
  const float* atom_w      = (const float*)d_in[6];
  const float* atom_b      = (const float*)d_in[7];
  const float* edge_w1     = (const float*)d_in[8];
  const float* edge_b1     = (const float*)d_in[9];
  const float* edge_w2     = (const float*)d_in[10];
  const float* edge_b2     = (const float*)d_in[11];
  const float* node_w1     = (const float*)d_in[12];
  const float* node_b1     = (const float*)d_in[13];
  const float* node_w2     = (const float*)d_in[14];
  const float* node_b2     = (const float*)d_in[15];
  const float* coord_w1    = (const float*)d_in[16];
  const float* coord_b1    = (const float*)d_in[17];
  const float* coord_w2    = (const float*)d_in[18];
  const float* ln_g        = (const float*)d_in[19];
  const float* ln_b        = (const float*)d_in[20];
  const float* head_w1     = (const float*)d_in[21];
  const float* head_b1     = (const float*)d_in[22];
  const float* head_w2     = (const float*)d_in[23];
  const float* head_b2     = (const float*)d_in[24];

  const int* row = edge_index;
  const int* col = edge_index + NEdg;

  size_t off = 0;
  auto carve = [&](size_t bytes) {
    char* p = (char*)d_ws + off;
    off += (bytes + 255) & ~(size_t)255;
    return p;
  };
  float*    h     = (float*)carve((size_t)NNod * HIDd * 4);
  _Float16* h16   = (_Float16*)carve((size_t)NNod * HIDd * 2);
  float*    agg   = (float*)carve((size_t)NNod * HIDd * 4);
  float*    upd   = (float*)carve((size_t)NNod * 3 * 4);
  float*    deg   = (float*)carve((size_t)NNod * 4);
  float*    pos   = (float*)carve((size_t)NNod * 3 * 4);
  _Float16* ew1_t = (_Float16*)carve((size_t)NLay * HIDd * 256 * 2);
  _Float16* ew2_t = (_Float16*)carve((size_t)NLay * HIDd * HIDd * 2);
  _Float16* nw1_t = (_Float16*)carve((size_t)NLay * HIDd * 256 * 2);
  _Float16* nw2_t = (_Float16*)carve((size_t)NLay * HIDd * HIDd * 2);
  _Float16* cw1_t = (_Float16*)carve((size_t)NLay * HIDd * HIDd * 2);
  _Float16* hw1_t = (_Float16*)carve((size_t)64 * HIDd * 2);

  // ---- weight conversion / transpose (f32 -> f16 [N][K]) ----
  for (int l = 0; l < NLay; ++l) {
    convert_wt<<<(256 * 128 + 255) / 256, 256, 0, stream>>>(
        edge_w1 + (size_t)l * 257 * 128, ew1_t + (size_t)l * 128 * 256, 256, 128);
    convert_wt<<<(128 * 128 + 255) / 256, 256, 0, stream>>>(
        edge_w2 + (size_t)l * 128 * 128, ew2_t + (size_t)l * 128 * 128, 128, 128);
    convert_wt<<<(256 * 128 + 255) / 256, 256, 0, stream>>>(
        node_w1 + (size_t)l * 256 * 128, nw1_t + (size_t)l * 128 * 256, 256, 128);
    convert_wt<<<(128 * 128 + 255) / 256, 256, 0, stream>>>(
        node_w2 + (size_t)l * 128 * 128, nw2_t + (size_t)l * 128 * 128, 128, 128);
    convert_wt<<<(128 * 128 + 255) / 256, 256, 0, stream>>>(
        coord_w1 + (size_t)l * 128 * 128, cw1_t + (size_t)l * 128 * 128, 128, 128);
  }
  convert_wt<<<(128 * 64 + 255) / 256, 256, 0, stream>>>(head_w1, hw1_t, 128, 64);

  // ---- degree (row is constant across layers) ----
  zero_kernel<<<(NNod + 255) / 256, 256, 0, stream>>>(deg, NNod);
  deg_kernel<<<(NEdg + 255) / 256, 256, 0, stream>>>(row, deg);

  // ---- node embedding + position init ----
  init_kernel<<<NNod, 128, 0, stream>>>(z, atom_types, latent_w, latent_b,
                                        atom_w, atom_b, init_coords, h, h16, pos);

  // ---- message-passing layers ----
  for (int l = 0; l < NLay; ++l) {
    zero_kernel<<<(int)(((size_t)NNod * HIDd + 255) / 256), 256, 0, stream>>>(
        agg, (size_t)NNod * HIDd);
    zero_kernel<<<(NNod * 3 + 255) / 256, 256, 0, stream>>>(upd, NNod * 3);
    edge_kernel<<<NEdg / 16, 256, 0, stream>>>(
        h16, pos, row, col,
        ew1_t + (size_t)l * 128 * 256, edge_w1 + (size_t)(l * 257 + 256) * 128,
        edge_b1 + l * 128,
        ew2_t + (size_t)l * 128 * 128, edge_b2 + l * 128,
        cw1_t + (size_t)l * 128 * 128, coord_b1 + l * 128, coord_w2 + l * 128,
        agg, upd);
    node_kernel<<<NNod / 16, 256, 0, stream>>>(
        h16, agg,
        nw1_t + (size_t)l * 128 * 256, node_b1 + l * 128,
        nw2_t + (size_t)l * 128 * 128, node_b2 + l * 128,
        ln_g + l * 128, ln_b + l * 128, h, h16);
    pos_update<<<(NNod * 3 + 255) / 256, 256, 0, stream>>>(pos, upd, deg);
  }

  // ---- output head ----
  head_kernel<<<NNod / 16, 256, 0, stream>>>(h16, pos, hw1_t, head_b1,
                                             head_w2, head_b2, (float*)d_out);
}
